// CondConv2d_3882650435883
// MI455X (gfx1250) — matrix operand order repaired and checked by reading, not compiled, not run on hardware
//
#include <hip/hip_runtime.h>
#include <hip/hip_bf16.h>

// ---- problem dims (compile-time; match reference) ----
#define BATCH   32
#define CIN     192
#define HH      56
#define WW      56
#define HW      3136          // 56*56
#define COUT    192
#define KDIM    1728          // CIN*3*3
#define KSTEPS  54            // KDIM/32
#define MTILES  12            // COUT/16
#define NTILES  196           // HW/16
#define WELEMS  331776        // COUT*KDIM (per-sample mixed weight elements)

typedef __attribute__((ext_vector_type(16))) __bf16 v16bf;
typedef __attribute__((ext_vector_type(8)))  float  v8f;

// float -> bf16 bits, round-to-nearest-even
__device__ __forceinline__ unsigned int f2bf_bits(float f) {
    unsigned int u = __float_as_uint(f);
    return (u + 0x7FFFu + ((u >> 16) & 1u)) >> 16;
}
__device__ __forceinline__ unsigned int pack_bf16x2(float lo, float hi) {
    return (f2bf_bits(lo) & 0xFFFFu) | (f2bf_bits(hi) << 16);
}

// -------------------------------------------------------------------------
// Kernel 1: mix expert weights -> per-sample bf16 weights stored directly in
// WMMA A-fragment order: wsA[b][mtile][kstep][lane(32)][elem(16)]  (bf16)
// A 16x32 bf16 layout (ISA 7.12.2): lane = g*16 + (m%16),
//   g = (kk/8)&1 ; elem j = (kk<16) ? kk%8 : 8+kk%8
// One thread per (m,k); loops over all 32 samples so expert_weight is read once.
// -------------------------------------------------------------------------
__global__ __launch_bounds__(256) void mix_weights_kernel(
    const float* __restrict__ routing,   // [32,8]
    const float* __restrict__ ew,        // [8, WELEMS]
    unsigned short* __restrict__ wsA)    // [32][WELEMS] bf16 bits, fragment order
{
    __shared__ float rw[BATCH * 8];
    const int tid = threadIdx.x;
    rw[tid] = routing[tid];              // 256 == 32*8 exactly
    __syncthreads();

    const int idx = blockIdx.x * 256 + tid;   // 0 .. WELEMS-1 (grid = 1296)
    const int m = idx / KDIM;
    const int k = idx - m * KDIM;

    float w8[8];
#pragma unroll
    for (int e = 0; e < 8; ++e) w8[e] = ew[(size_t)e * WELEMS + idx];

    const int mtile = m >> 4, mrow = m & 15;
    const int kstep = k >> 5, kk   = k & 31;
    const int g = (kk >> 3) & 1;
    const int j = (kk < 16) ? (kk & 7) : (8 + (kk & 7));
    const size_t dst = ((size_t)mtile * KSTEPS + kstep) * 512
                     + (size_t)(g * 16 + mrow) * 16 + j;

#pragma unroll
    for (int b = 0; b < BATCH; ++b) {
        float s = 0.f;
#pragma unroll
        for (int e = 0; e < 8; ++e) s = fmaf(rw[b * 8 + e], w8[e], s);
        wsA[(size_t)b * WELEMS + dst] = (unsigned short)f2bf_bits(s);
    }
}

// -------------------------------------------------------------------------
// Kernel 2: mix bias -> wsBias[b][cout] (f32)
// -------------------------------------------------------------------------
__global__ __launch_bounds__(256) void mix_bias_kernel(
    const float* __restrict__ routing,   // [32,8]
    const float* __restrict__ eb,        // [8,192]
    float* __restrict__ wsBias)          // [32,192]
{
    const int idx = blockIdx.x * 256 + threadIdx.x;  // 0..6143 (grid = 24)
    const int b = idx / COUT;
    const int c = idx - b * COUT;
    float s = 0.f;
#pragma unroll
    for (int e = 0; e < 8; ++e) s = fmaf(routing[b * 8 + e], eb[e * COUT + c], s);
    wsBias[idx] = s;
}

// -------------------------------------------------------------------------
// Kernel 3: implicit-GEMM conv via WMMA bf16.
// Block = 128 threads (4 waves) handles one (sample b, N-tile of 16 spatial
// positions). Wave w computes M-tiles {w, w+4, w+8}. K loop: 54 steps of 32.
// Per K-step the block cooperatively builds the shared B fragment (im2col,
// 32x16 bf16 = 1 KB) in a DOUBLE-BUFFERED LDS slab (single barrier per step:
// passing barrier ks+1 proves all iter-ks reads of buffer ks&1 completed,
// so iter ks+2 may overwrite it). Each wave ds_loads its 32B v16bf and issues
// 3 WMMAs against 3 A fragments streamed (and prefetched) from the
// pre-laid-out ws buffer.
// B 32x16 bf16 layout assumed: lane = g*16+n, elem j -> K = (j<8) ? g*8+j : 16+g*8+(j-8).
// EXEC is all-ones at every WMMA site (no early returns; OOB -> select 0).
// -------------------------------------------------------------------------
__global__ __launch_bounds__(128) void cond_conv_wmma_kernel(
    const float* __restrict__ x,             // [32,192,56,56]
    const unsigned short* __restrict__ wsA,  // mixed weights, fragment order
    const float* __restrict__ wsBias,        // [32,192]
    float* __restrict__ out)                 // [32,192,56,56]
{
    const int b     = blockIdx.y;
    const int ntile = blockIdx.x;
    const int tid   = threadIdx.x;
    const int wave  = tid >> 5;
    const int lane  = tid & 31;

    // double-buffered B fragment: 2 x 2 KB, 1 KB of each used
    __shared__ __align__(32) unsigned int lds_b[2][512];

    const float* __restrict__ xb = x + (size_t)b * CIN * HW;

    // cooperative-B gather setup: thread t fills bf16 elems [4t .. 4t+3]
    // element index = blane*16 + e  ->  u32 slots 2t, 2t+1
    const int blane = tid >> 2;
    const int be0   = (tid & 3) * 4;
    const int bn    = ntile * 16 + (blane & 15);
    const int oh    = bn / WW;
    const int ow    = bn - oh * WW;
    const int kgrp  = (blane >> 4) * 8;

    v8f acc[3];
#pragma unroll
    for (int i = 0; i < 3; ++i)
        acc[i] = (v8f){0.f, 0.f, 0.f, 0.f, 0.f, 0.f, 0.f, 0.f};

    const unsigned short* aptr[3];
#pragma unroll
    for (int i = 0; i < 3; ++i) {
        const int mtile = wave + i * 4;
        aptr[i] = wsA + ((size_t)b * MTILES + mtile) * (KSTEPS * 512) + lane * 16;
    }

#pragma unroll 2
    for (int ks = 0; ks < KSTEPS; ++ks) {
        unsigned int* __restrict__ buf = lds_b[ks & 1];

        // ---- build im2col B fragment (4 elems per thread) ----
        unsigned int p0, p1;
        {
            const int kbase = ks * 32 + kgrp + be0 + (be0 & 8);
            float v[4];
#pragma unroll
            for (int jj = 0; jj < 4; ++jj) {
                const int k   = kbase + jj;
                const int cin = k / 9;
                const int r9  = k - cin * 9;
                const int kh  = r9 / 3;
                const int kw  = r9 - kh * 3;
                const int ih  = oh + kh - 1;
                const int iw  = ow + kw - 1;
                const bool ok = ((unsigned)ih < HH) && ((unsigned)iw < WW);
                v[jj] = ok ? xb[cin * HW + ih * WW + iw] : 0.0f;
            }
            p0 = pack_bf16x2(v[0], v[1]);
            p1 = pack_bf16x2(v[2], v[3]);
        }
        buf[2 * tid]     = p0;
        buf[2 * tid + 1] = p1;

        // prefetch next K-step's A fragments while this step's B settles
        if (ks + 1 < KSTEPS) {
#pragma unroll
            for (int i = 0; i < 3; ++i)
                __builtin_prefetch(aptr[i] + (size_t)(ks + 1) * 512, 0, 0);
        }

        __syncthreads();                       // fragment complete

        const v16bf bfrag = *reinterpret_cast<const v16bf*>(&buf[lane * 8]);

#pragma unroll
        for (int i = 0; i < 3; ++i) {
            const v16bf afrag =
                *reinterpret_cast<const v16bf*>(aptr[i] + (size_t)ks * 512);
            acc[i] = __builtin_amdgcn_wmma_f32_16x16x32_bf16(
                false, afrag, false, bfrag, (short)0, acc[i], false, false);
        }
    }

    // ---- epilogue: C/D 16x16 f32 layout: reg r, lane l -> M = (l/16)*8 + r,
    // N = l%16. Output is write-once: use non-temporal stores to preserve the
    // L2-resident x / mixed-weight working set. ----
    const int nout  = ntile * 16 + (lane & 15);
    const int mbase = (lane >> 4) * 8;
#pragma unroll
    for (int i = 0; i < 3; ++i) {
        const int mtile = wave + i * 4;
        const float* __restrict__ bp = wsBias + b * COUT + mtile * 16 + mbase;
        float* __restrict__ op =
            out + ((size_t)b * COUT + mtile * 16 + mbase) * HW + nout;
#pragma unroll
        for (int r = 0; r < 8; ++r) {
            __builtin_nontemporal_store(acc[i][r] + bp[r], op + (size_t)r * HW);
        }
    }
}

// -------------------------------------------------------------------------
extern "C" void kernel_launch(void* const* d_in, const int* in_sizes, int n_in,
                              void* d_out, int out_size, void* d_ws, size_t ws_size,
                              hipStream_t stream) {
    (void)in_sizes; (void)n_in; (void)out_size; (void)ws_size;
    const float* x       = (const float*)d_in[0];   // [32,192,56,56]
    const float* routing = (const float*)d_in[1];   // [32,8]
    const float* ew      = (const float*)d_in[2];   // [8, 331776]
    const float* eb      = (const float*)d_in[3];   // [8, 192]
    float*       out     = (float*)d_out;

    // ws layout: [0, 21233664) bf16 mixed weights (fragment order),
    //            then [32,192] f32 mixed bias.
    unsigned short* wsA    = (unsigned short*)d_ws;
    float*          wsBias = (float*)((char*)d_ws + (size_t)BATCH * WELEMS * 2);

    mix_weights_kernel<<<WELEMS / 256, 256, 0, stream>>>(routing, ew, wsA);
    mix_bias_kernel<<<(BATCH * COUT) / 256, 256, 0, stream>>>(routing, eb, wsBias);

    dim3 grid(NTILES, BATCH);
    cond_conv_wmma_kernel<<<grid, 128, 0, stream>>>(x, wsA, wsBias, out);
}
